// SCNN_Model_LIF_hh_88218628260287
// MI455X (gfx1250) — matrix operand order repaired and checked by reading, not compiled, not run on hardware
//
#include <hip/hip_runtime.h>
#include <hip/hip_bf16.h>

typedef _Float16 half_t;
typedef __attribute__((ext_vector_type(16))) _Float16 v16h;
typedef __attribute__((ext_vector_type(8)))  _Float16 v8h;
typedef __attribute__((ext_vector_type(8)))  float    v8f;

#define B_      128
#define H1      36
#define H2      18
#define NCH     10
#define N2      (B_*H2*H2)         /* 41472 GEMM columns for conv2 */
#define FCK     3240
#define FCKP    3264               /* FCK padded to 32 */
#define THRESH_ 0.3f
#define DECAY_  0.2f

/* ---- workspace layout ---- */
/* float region */
#define SZ_C1M  (B_*NCH*H1*H1*4)   /* 6,635,520 */
#define SZ_C2M  (B_*NCH*H2*H2*4)   /* 1,658,880 */
#define SZ_HSUM (B_*FCK)           /*   414,720 */
#define SZ_Y2   (30*N2)            /* 1,244,160 */
#define O_C1M   0
#define O_C1S   (O_C1M + SZ_C1M)
#define O_C2M   (O_C1S + SZ_C1M)
#define O_C2S   (O_C2M + SZ_C2M)
#define O_HSUM  (O_C2S + SZ_C2M)
#define O_Y2    (O_HSUM + SZ_HSUM)
#define O_OUTS  (O_Y2 + SZ_Y2)
#define O_HBASE (O_OUTS + B_*50)   /* 18,254,080 floats; 32B aligned */
#define ZTOT    O_Y2               /* zero c1m,c1s,c2m,c2s,hsum */
/* half region (offsets in half_t units, all 32B aligned) */
#define HO_X2   0                  /* f16 pooled spikes  [B,10,18,18]   */
#define HO_BT   (HO_X2 + B_*NCH*H2*H2)        /* im2col panel [N2][96]  */
#define HO_AW   (HO_BT + N2*96)               /* conv2 weights [32][96] */
#define HO_HS   (HO_AW + 32*96)               /* hsum/15 f16 [128][FCKP]*/
#define HO_FCW  (HO_HS + B_*FCKP)             /* fc_w f16 [64][FCKP]    */

__device__ inline v16h cat8(v8h lo, v8h hi) {
    return __builtin_shufflevector(lo, hi, 0,1,2,3,4,5,6,7,8,9,10,11,12,13,14,15);
}

__global__ void zero_ws(float* ws) {
    int i = blockIdx.x * blockDim.x + threadIdx.x;
    if (i < ZTOT) ws[i] = 0.f;
}

/* padded f16 conv2 weight matrix Aw[32][96]; w2 flat [3,10,10,3,3] => m*90+k */
__global__ void build_aw(const float* w2, half_t* aw) {
    int i = blockIdx.x * blockDim.x + threadIdx.x;
    if (i >= 32 * 96) return;
    int m = i / 96, k = i % 96;
    aw[i] = (m < 30 && k < 90) ? (half_t)w2[m * 90 + k] : (half_t)0.f;
}

/* padded f16 FC weights [64][FCKP]; fc_w is [50][3240] row-major (= [col][k]) */
__global__ void build_fcw(const float* fc_w, half_t* fcw) {
    int i = blockIdx.x * blockDim.x + threadIdx.x;
    if (i >= 64 * FCKP) return;
    int c = i / FCKP, k = i % FCKP;
    fcw[i] = (c < 50 && k < FCK) ? (half_t)fc_w[c * FCK + k] : (half_t)0.f;
}

/* f16 A-panel for FC: hs[128][FCKP] = hsum/15 */
__global__ void build_hs(const float* __restrict__ hsum, half_t* __restrict__ hs) {
    int i = blockIdx.x * blockDim.x + threadIdx.x;
    if (i >= B_ * FCKP) return;
    int m = i / FCKP, k = i % FCKP;
    hs[i] = (k < FCK) ? (half_t)(hsum[m * FCK + k] * (1.0f / 15.0f)) : (half_t)0.f;
}

/* conv1 (1->30ch, 3x3 pad1) + inner drive + LIF update of c1m/c1s */
__global__ void conv1_lif(const float* __restrict__ x, const float* __restrict__ w1,
                          const float* __restrict__ b1, const float* __restrict__ fw,
                          const float* __restrict__ fb, float* c1m, float* c1s) {
    int idx = blockIdx.x * blockDim.x + threadIdx.x;
    const int TOT = B_ * NCH * H1 * H1;
    if (idx >= TOT) return;
    int w = idx % H1, t = idx / H1;
    int h = t % H1;  t /= H1;
    int c = t % NCH; int b = t / NCH;

    float patch[9];
    #pragma unroll
    for (int dy = 0; dy < 3; ++dy)
        #pragma unroll
        for (int dx = 0; dx < 3; ++dx) {
            int hh = h + dy - 1, ww = w + dx - 1;
            patch[dy*3+dx] = ((unsigned)hh < (unsigned)H1 && (unsigned)ww < (unsigned)H1)
                           ? x[(b*H1 + hh)*H1 + ww] : 0.f;
        }
    float drive[4];
    #pragma unroll
    for (int g = 0; g < 3; ++g) {
        float acc = b1[g*NCH + c];
        const float* wg = w1 + (g*NCH + c)*9;
        #pragma unroll
        for (int kk = 0; kk < 9; ++kk) acc += patch[kk] * wg[kk];
        drive[g] = acc;
    }
    int base = idx * 4;
    float m0 = c1m[base], m1 = c1m[base+1], m2 = c1m[base+2], m3 = c1m[base+3];
    drive[3] = fb[0] + fw[0]*m0 + fw[1]*m1 + fw[2]*m2;  /* uses OLD mem */
    float mm[4] = {m0, m1, m2, m3};
    #pragma unroll
    for (int s = 0; s < 4; ++s) {
        float sp = c1s[base + s];
        float nm = mm[s] * DECAY_ * (1.f - sp) + drive[s];
        c1m[base + s] = nm;
        c1s[base + s] = nm > THRESH_ ? 1.f : 0.f;
    }
}

/* x2h[b,c,y,x] (f16) = sum over 4 pathways of 2x2 avg-pooled c1s (exact in f16) */
__global__ void pool1_sum(const float* __restrict__ c1s, half_t* __restrict__ x2h) {
    int idx = blockIdx.x * blockDim.x + threadIdx.x;
    const int TOT = B_ * NCH * H2 * H2;
    if (idx >= TOT) return;
    int xx = idx % H2, t = idx / H2;
    int yy = t % H2;  t /= H2;
    int c = t % NCH;  int b = t / NCH;
    float s = 0.f;
    #pragma unroll
    for (int dy = 0; dy < 2; ++dy)
        #pragma unroll
        for (int dx = 0; dx < 2; ++dx) {
            int base = (((b*NCH + c)*H1 + 2*yy + dy)*H1 + 2*xx + dx) * 4;
            s += c1s[base] + c1s[base+1] + c1s[base+2] + c1s[base+3];
        }
    x2h[idx] = (half_t)(0.25f * s);
}

/* im2col staging: bt[n][96] so each lane's 16 B-fragment halves are two contiguous 8-half runs */
__global__ void build_bt(const half_t* __restrict__ x2h, half_t* __restrict__ bt) {
    int n = blockIdx.x * blockDim.x + threadIdx.x;
    if (n >= N2) return;
    int b = n / (H2*H2), rem = n % (H2*H2);
    int y = rem / H2, x = rem % H2;
    half_t tmp[96];
    #pragma unroll
    for (int ci = 0; ci < NCH; ++ci) {
        const half_t* src = x2h + (b*NCH + ci) * (H2*H2);
        #pragma unroll
        for (int dy = 0; dy < 3; ++dy) {
            int h = y + dy - 1;
            bool okh = (unsigned)h < (unsigned)H2;
            #pragma unroll
            for (int dx = 0; dx < 3; ++dx) {
                int w = x + dx - 1;
                bool ok = okh && ((unsigned)w < (unsigned)H2);
                tmp[ci*9 + dy*3 + dx] = ok ? src[h*H2 + w] : (half_t)0.f;
            }
        }
    }
    #pragma unroll
    for (int k = 90; k < 96; ++k) tmp[k] = (half_t)0.f;
    v8h* dst = (v8h*)(bt + n * 96);
    #pragma unroll
    for (int j = 0; j < 12; ++j) {
        v8h v;
        #pragma unroll
        for (int e = 0; e < 8; ++e) v[e] = tmp[8*j + e];
        dst[j] = v;
    }
}

/* conv2 implicit GEMM: C[32 x 16-col tile] = Aw[32x96] * Bt^T ; pure WMMA kernel */
__global__ void conv2_wmma(const half_t* __restrict__ bt, const half_t* __restrict__ aw,
                           float* __restrict__ y2) {
    int lane = threadIdx.x & 31;
    int tile = blockIdx.x * (blockDim.x >> 5) + (threadIdx.x >> 5); /* wave-uniform */
    int hf = lane >> 4;
    int lm = lane & 15;
    int ncol = tile * 16 + lm;
    const half_t* bp = bt + ncol * 96;
    __builtin_prefetch(bp, 0, 3);                 /* global_prefetch_b8 */
    const half_t* a0p = aw + lm * 96;
    const half_t* a1p = aw + (lm + 16) * 96;

    v8f acc0 = {}; v8f acc1 = {};
    #pragma unroll
    for (int it = 0; it < 3; ++it) {
        int kb = it * 32;
        v16h bf = cat8(*(const v8h*)(bp + kb + hf*8),              /* 16B aligned */
                       *(const v8h*)(bp + kb + 16 + hf*8));
        v16h a0 = cat8(*(const v8h*)(a0p + kb + hf*8),
                       *(const v8h*)(a0p + kb + 16 + hf*8));
        v16h a1 = cat8(*(const v8h*)(a1p + kb + hf*8),
                       *(const v8h*)(a1p + kb + 16 + hf*8));
        acc0 = __builtin_amdgcn_wmma_f32_16x16x32_f16(false, a0, false, bf, (short)0, acc0, false, false);
        acc1 = __builtin_amdgcn_wmma_f32_16x16x32_f16(false, a1, false, bf, (short)0, acc1, false, false);
    }
    #pragma unroll
    for (int j = 0; j < 8; ++j) {
        int m0 = j + hf*8;            /* rows 0..15  */
        y2[m0*N2 + ncol] = acc0[j];
        int m1 = 16 + j + hf*8;       /* rows 16..31, 30/31 are padding */
        if (m1 < 30) y2[m1*N2 + ncol] = acc1[j];
    }
}

/* layer-2 LIF: drive = conv out + bias, inner from old c2m; update c2m/c2s */
__global__ void lif2(const float* __restrict__ y2, const float* __restrict__ b2,
                     const float* __restrict__ fw, const float* __restrict__ fb,
                     float* c2m, float* c2s) {
    int idx = blockIdx.x * blockDim.x + threadIdx.x;
    const int TOT = B_ * NCH * H2 * H2;
    if (idx >= TOT) return;
    int pos = idx % (H2*H2);
    int co  = (idx / (H2*H2)) % NCH;
    int b   = idx / (NCH*H2*H2);
    int n = b * (H2*H2) + pos;
    float drive[4];
    #pragma unroll
    for (int g = 0; g < 3; ++g)
        drive[g] = y2[(g*NCH + co)*N2 + n] + b2[g*NCH + co];
    int base = idx * 4;
    float m0 = c2m[base], m1 = c2m[base+1], m2 = c2m[base+2], m3 = c2m[base+3];
    drive[3] = fb[0] + fw[0]*m0 + fw[1]*m1 + fw[2]*m2;
    float mm[4] = {m0, m1, m2, m3};
    #pragma unroll
    for (int s = 0; s < 4; ++s) {
        float sp = c2s[base + s];
        float nm = mm[s] * DECAY_ * (1.f - sp) + drive[s];
        c2m[base + s] = nm;
        c2s[base + s] = nm > THRESH_ ? 1.f : 0.f;
    }
}

/* hsum += flatten(2x2 avg pool of c2s) ; flat index ((c*9+yy)*9+xx)*4+s */
__global__ void pool2_acc(const float* __restrict__ c2s, float* __restrict__ hsum) {
    int idx = blockIdx.x * blockDim.x + threadIdx.x;
    const int TOT = B_ * FCK;
    if (idx >= TOT) return;
    int f = idx % FCK, b = idx / FCK;
    int s = f % 4, r = f / 4;
    int xx = r % 9; r /= 9;
    int yy = r % 9; int c = r / 9;
    float acc = 0.f;
    #pragma unroll
    for (int dy = 0; dy < 2; ++dy)
        #pragma unroll
        for (int dx = 0; dx < 2; ++dx)
            acc += c2s[(((b*NCH + c)*H2 + 2*yy + dy)*H2 + 2*xx + dx)*4 + s];
    hsum[idx] += 0.25f * acc;
}

/* outs[128,50] = hs16 @ fcw16^T + fc_b ; one wave per 16x16 tile, pure WMMA */
__global__ void fc_wmma(const half_t* __restrict__ hs, const half_t* __restrict__ fcw,
                        const float* __restrict__ fc_b, float* __restrict__ outs) {
    int lane = threadIdx.x;
    int mt = blockIdx.x >> 2, nt = blockIdx.x & 3;   /* 8 x 4 tiles (N padded to 64) */
    int lm = lane & 15, hf = lane >> 4;
    int row = mt * 16;
    int col = nt * 16 + lm;
    const half_t* ap = hs + (row + lm) * FCKP;
    const half_t* bp = fcw + col * FCKP;
    __builtin_prefetch(ap, 0, 3);
    v8f acc = {};
    for (int it = 0; it < FCKP / 32; ++it) {         /* 102 iterations */
        int kb = it * 32;
        v16h a  = cat8(*(const v8h*)(ap + kb + hf*8),
                       *(const v8h*)(ap + kb + 16 + hf*8));
        v16h bf = cat8(*(const v8h*)(bp + kb + hf*8),
                       *(const v8h*)(bp + kb + 16 + hf*8));
        acc = __builtin_amdgcn_wmma_f32_16x16x32_f16(false, a, false, bf, (short)0, acc, false, false);
    }
    if (col < 50) {
        #pragma unroll
        for (int j = 0; j < 8; ++j) {
            int m = row + j + hf*8;
            outs[m*50 + col] = acc[j] + fc_b[col];
        }
    }
}

/* out[b,t,o] = sum_f outs[b,f]*task_w[t,o,f] + task_b[t,o] */
__global__ void task_head(const float* __restrict__ outs, const float* __restrict__ tw,
                          const float* __restrict__ tb, float* __restrict__ out) {
    int idx = blockIdx.x * blockDim.x + threadIdx.x;
    if (idx >= B_ * 100) return;
    int o = idx % 10, t = (idx / 10) % 10, b = idx / 100;
    float acc = tb[t*10 + o];
    const float* w = tw + (t*10 + o) * 50;
    const float* v = outs + b * 50;
    #pragma unroll
    for (int f = 0; f < 50; ++f) acc += v[f] * w[f];
    out[idx] = acc;  /* idx = b*100 + t*10 + o == [B,T,10] flat */
}

extern "C" void kernel_launch(void* const* d_in, const int* in_sizes, int n_in,
                              void* d_out, int out_size, void* d_ws, size_t ws_size,
                              hipStream_t stream) {
    (void)in_sizes; (void)n_in; (void)out_size; (void)ws_size;
    const float* x     = (const float*)d_in[0];
    const float* w1    = (const float*)d_in[1];
    const float* b1    = (const float*)d_in[2];
    const float* w2    = (const float*)d_in[3];
    const float* b2    = (const float*)d_in[4];
    const float* fw    = (const float*)d_in[5];
    const float* fb    = (const float*)d_in[6];
    const float* fc_w  = (const float*)d_in[7];
    const float* fc_b  = (const float*)d_in[8];
    const float* tw    = (const float*)d_in[9];
    const float* tb    = (const float*)d_in[10];
    float* ws   = (float*)d_ws;
    float* out  = (float*)d_out;

    float* c1m   = ws + O_C1M;
    float* c1s   = ws + O_C1S;
    float* c2m   = ws + O_C2M;
    float* c2s   = ws + O_C2S;
    float* hsum  = ws + O_HSUM;
    float* y2    = ws + O_Y2;
    float* outs  = ws + O_OUTS;
    half_t* hbase = (half_t*)(ws + O_HBASE);
    half_t* x2h  = hbase + HO_X2;
    half_t* bt   = hbase + HO_BT;
    half_t* aw   = hbase + HO_AW;
    half_t* hs16 = hbase + HO_HS;
    half_t* fcw16= hbase + HO_FCW;

    zero_ws<<<(ZTOT + 255)/256, 256, 0, stream>>>(ws);
    build_aw<<<12, 256, 0, stream>>>(w2, aw);
    build_fcw<<<(64*FCKP + 255)/256, 256, 0, stream>>>(fc_w, fcw16);

    const int TOT1 = B_*NCH*H1*H1;   /* 1,658,880 */
    const int TOT2 = B_*NCH*H2*H2;   /*   414,720 */
    for (int step = 0; step < 15; ++step) {
        conv1_lif<<<(TOT1 + 255)/256, 256, 0, stream>>>(x, w1, b1, fw, fb, c1m, c1s);
        pool1_sum<<<(TOT2 + 255)/256, 256, 0, stream>>>(c1s, x2h);
        build_bt<<<(N2 + 255)/256, 256, 0, stream>>>(x2h, bt);
        conv2_wmma<<<N2/16/4, 128, 0, stream>>>(bt, aw, y2);   /* 648 blocks x 4 waves */
        lif2<<<(TOT2 + 255)/256, 256, 0, stream>>>(y2, b2, fw, fb, c2m, c2s);
        pool2_acc<<<(B_*FCK + 255)/256, 256, 0, stream>>>(c2s, hsum);
    }
    build_hs<<<(B_*FCKP + 255)/256, 256, 0, stream>>>(hsum, hs16);
    fc_wmma<<<32, 32, 0, stream>>>(hs16, fcw16, fc_b, outs);
    task_head<<<(B_*100 + 255)/256, 256, 0, stream>>>(outs, tw, tb, out);
}
